// TransfoModel_49967649521834
// MI455X (gfx1250) — compile-verified, matrix operand order checked by reading, not run on hardware
//
#include <hip/hip_runtime.h>
#include <hip/hip_bf16.h>
#include <math.h>

typedef __bf16 bf16;
typedef __attribute__((ext_vector_type(16))) __bf16 v16bf;
typedef __attribute__((ext_vector_type(8)))  float  v8f;

#define BB 8
#define LL 400
#define FF 128
#define DD 256
#define TT 256
#define HH 512
#define NHH 8
#define HDD 64
#define PPOS 1200
#define MM (BB*LL)      // 3200
#define NDELTA 799      // deltas d = l-m in [-399,399]

// ---------------------------------------------------------------- helpers

__device__ __forceinline__ v8f wmma_bf16(v16bf a, v16bf b, v8f c) {
  return __builtin_amdgcn_wmma_f32_16x16x32_bf16(false, a, false, b, (short)0, c, false, false);
}

// A/B fragment: element e -> row = row0 + (lane&15), k = k0 + (e/8)*16 + (lane>>4)*8 + e%8
__device__ __forceinline__ v16bf frag_load(const bf16* __restrict__ base, int stride, int row0, int k0) {
  int lane = threadIdx.x & 31;
  const bf16* p = base + (size_t)(row0 + (lane & 15)) * stride + k0 + ((lane >> 4) << 3);
  v16bf f;
#pragma unroll
  for (int e = 0; e < 8; ++e) f[e] = p[e];
#pragma unroll
  for (int e = 0; e < 8; ++e) f[8 + e] = p[16 + e];
  return f;
}

__device__ __forceinline__ v16bf frag_load_guard(const bf16* __restrict__ base, int stride, int row0, int k0, int rowMax) {
  int lane = threadIdx.x & 31;
  int r = row0 + (lane & 15);
  v16bf f;
  if (r < rowMax) {
    const bf16* p = base + (size_t)r * stride + k0 + ((lane >> 4) << 3);
#pragma unroll
    for (int e = 0; e < 8; ++e) f[e] = p[e];
#pragma unroll
    for (int e = 0; e < 8; ++e) f[8 + e] = p[16 + e];
  } else {
#pragma unroll
    for (int e = 0; e < 16; ++e) f[e] = (bf16)0.0f;
  }
  return f;
}

// B fragment from row-major [K, ...] matrix: B[k,n] = base[k*stride + col0 + (lane&15)], k guarded
__device__ __forceinline__ v16bf frag_load_t(const bf16* __restrict__ base, int stride, int k0, int col0, int kMax) {
  int lane = threadIdx.x & 31;
  const bf16* p = base + col0 + (lane & 15);
  int kk = k0 + ((lane >> 4) << 3);
  v16bf f;
#pragma unroll
  for (int e = 0; e < 8; ++e)  { int k = kk + e;      f[e]     = (k < kMax) ? p[(size_t)k * stride] : (bf16)0.0f; }
#pragma unroll
  for (int e = 0; e < 8; ++e)  { int k = kk + 16 + e; f[8 + e] = (k < kMax) ? p[(size_t)k * stride] : (bf16)0.0f; }
  return f;
}

__device__ __forceinline__ float mishf(float x) {
  float sp = (x > 20.f) ? x : log1pf(__expf(x));
  return x * tanhf(sp);
}

// im2col fragment for the 3-tap 'SAME' conv: logical A[m, kk], kk = dl*256 + c,
// value = f1[b, l+dl-1, c] with zero padding. f1 lives in featsb columns 0..255 (ld=512).
__device__ __forceinline__ v16bf conv_frag(const bf16* __restrict__ featsb, int row0, int ks) {
  int lane = threadIdx.x & 31;
  int m = row0 + (lane & 15);
  int bi = m / LL, l = m - bi * LL;
  int khi = (lane >> 4) << 3;
  v16bf f;
#pragma unroll
  for (int g = 0; g < 2; ++g) {
    int kb0 = ks + khi + g * 16;
    int dl = kb0 >> 8;
    int c  = kb0 & 255;
    int l2 = l + dl - 1;
    if (l2 >= 0 && l2 < LL) {
      const bf16* p = featsb + ((size_t)(bi * LL + l2)) * HH + c;
#pragma unroll
      for (int e = 0; e < 8; ++e) f[g * 8 + e] = p[e];
    } else {
#pragma unroll
      for (int e = 0; e < 8; ++e) f[g * 8 + e] = (bf16)0.0f;
    }
  }
  return f;
}

// ---------------------------------------------------------------- small prep kernels

__global__ void prep_kernel(const float* __restrict__ x, bf16* __restrict__ xb, float* __restrict__ masks) {
  int i = blockIdx.x * 256 + threadIdx.x;
  if (i < MM * FF) xb[i] = (bf16)x[i];
  if (i < 4 * BB * LL) {
    int o = i / (BB * LL);
    int r = i - o * (BB * LL);
    int b = r / LL, l = r - b * LL;
    const float* xr = x + ((size_t)b * LL + l) * FF;
    float v;
    if (o == 0)      v = xr[4];
    else if (o == 1) v = fmaxf(xr[2], xr[3]);
    else if (o == 2) v = xr[0];
    else             v = xr[1];
    masks[i] = v;
  }
}

__device__ __forceinline__ int bpos_dev(int rel) {
  const int mid = 300;
  int sgn = (rel > 0) - (rel < 0);
  int apos = (rel < mid && rel > -mid) ? (mid - 1) : (rel < 0 ? -rel : rel);
  double logp = ceil(log((double)apos / (double)mid) / log(599.0 / 300.0) * (double)(mid - 1)) + (double)mid;
  return (apos <= mid) ? rel : (int)((double)sgn * logp);
}

__global__ void idx_kernel(int* __restrict__ c2pT, int* __restrict__ p2cT) {
  int dd = blockIdx.x * 256 + threadIdx.x;
  if (dd >= NDELTA) return;
  int d = dd - 399;
  int b1 = bpos_dev(d) + 600;              // c2p index for rel = d
  b1 = b1 < 0 ? 0 : (b1 > 1199 ? 1199 : b1);
  int b2 = -bpos_dev(-d) + 600;            // p2c index for rel = -d (i.e. p2c_pos[m,l], d=l-m)
  b2 = b2 < 0 ? 0 : (b2 > 1199 ? 1199 : b2);
  c2pT[dd] = b1;
  p2cT[dd] = b2;
}

__global__ void pack_wt(const float* __restrict__ W, bf16* __restrict__ WT, int K, int N) {
  int i = blockIdx.x * 256 + threadIdx.x;
  if (i >= K * N) return;
  int k = i / N, n = i - k * N;
  WT[(size_t)n * K + k] = (bf16)W[i];
}

__global__ void gather_pos(const bf16* __restrict__ pkb, const bf16* __restrict__ pqb,
                           const int* __restrict__ c2pT, const int* __restrict__ p2cT,
                           bf16* __restrict__ PKg, bf16* __restrict__ PQg) {
  int i = blockIdx.x * 256 + threadIdx.x;
  if (i >= NHH * NDELTA * HDD) return;
  int h = i / (NDELTA * HDD);
  int r = i - h * (NDELTA * HDD);
  int dd = r / HDD, d = r - dd * HDD;
  PKg[i] = pkb[(size_t)c2pT[dd] * HH + h * HDD + d];
  PQg[i] = pqb[(size_t)p2cT[dd] * HH + h * HDD + d];
}

// row-wise LayerNorm: one block per row, biased variance, eps=1e-7
__global__ void __launch_bounds__(256) ln_kernel(const float* __restrict__ X,
    const float* __restrict__ g, const float* __restrict__ bta,
    float* __restrict__ Yf, bf16* __restrict__ Yb, int N) {
  int row = blockIdx.x;
  const float* xr = X + (size_t)row * N;
  __shared__ float red[256];
  float s = 0.f, s2 = 0.f;
  for (int c = threadIdx.x; c < N; c += 256) { float v = xr[c]; s += v; s2 += v * v; }
  red[threadIdx.x] = s; __syncthreads();
  for (int o = 128; o; o >>= 1) { if (threadIdx.x < o) red[threadIdx.x] += red[threadIdx.x + o]; __syncthreads(); }
  float mean = red[0] / (float)N; __syncthreads();
  red[threadIdx.x] = s2; __syncthreads();
  for (int o = 128; o; o >>= 1) { if (threadIdx.x < o) red[threadIdx.x] += red[threadIdx.x + o]; __syncthreads(); }
  float var = red[0] / (float)N - mean * mean;
  float inv = rsqrtf(var + 1e-7f);
  for (int c = threadIdx.x; c < N; c += 256) {
    float v = (xr[c] - mean) * inv * g[c] + bta[c];
    if (Yf) Yf[(size_t)row * N + c] = v;
    if (Yb) Yb[(size_t)row * N + c] = (bf16)v;
  }
}

// ---------------------------------------------------------------- generic bf16 WMMA GEMM
// C[M,N] = act(A[M,K] @ WT[N,K]^T + bias [+ R])
// block = 8 waves arranged 4(M) x 2(N) -> 64 x 128 block tile; wave = 16(M) x 64(N)
// One A fragment is reused across 4 B fragments: ~32 MACs/byte of L2 traffic.
template<int CONV, int ACT>
__global__ void __launch_bounds__(256) gemm_wmma(
    const bf16* __restrict__ A, int lda,
    const bf16* __restrict__ WT, const float* __restrict__ bias,
    const float* __restrict__ R, int ldr,
    float* __restrict__ Cf, int ldcf, bf16* __restrict__ Cb, int ldcb,
    int M, int N, int K)
{
  int wid = threadIdx.x >> 5, lane = threadIdx.x & 31;
  int row0 = blockIdx.x * 64 + (wid >> 1) * 16;
  int col0 = blockIdx.y * 128 + (wid & 1) * 64;
  if (row0 >= M) return;                      // wave-uniform
  v8f acc[4] = {};
  for (int ks = 0; ks < K; ks += 32) {
    v16bf a = CONV ? conv_frag(A, row0, ks) : frag_load(A, lda, row0, ks);
    if (!CONV && (ks + 32 < K))
      __builtin_prefetch(A + (size_t)(row0 + (lane & 15)) * lda + ks + 32, 0, 0);
    acc[0] = wmma_bf16(a, frag_load(WT, K, col0 +  0, ks), acc[0]);
    acc[1] = wmma_bf16(a, frag_load(WT, K, col0 + 16, ks), acc[1]);
    acc[2] = wmma_bf16(a, frag_load(WT, K, col0 + 32, ks), acc[2]);
    acc[3] = wmma_bf16(a, frag_load(WT, K, col0 + 48, ks), acc[3]);
  }
  int ib = (lane >> 4) * 8;
#pragma unroll
  for (int nt = 0; nt < 4; ++nt) {
    int n = col0 + nt * 16 + (lane & 15);
    float bv = bias ? bias[n] : 0.0f;
#pragma unroll
    for (int v = 0; v < 8; ++v) {
      int row = row0 + ib + v;
      float val = acc[nt][v] + bv;
      if (R) val += R[(size_t)row * ldr + n];
      if (ACT == 1) val = mishf(val);
      else if (ACT == 2) val = fmaxf(val, 0.0f);
      if (Cf) Cf[(size_t)row * ldcf + n] = val;
      if (Cb) Cb[(size_t)row * ldcb + n] = (bf16)val;
    }
  }
}

// ---------------------------------------------------------------- disentangled attention
// grid = (25 row-blocks, B*NH); block = 256 threads (8 waves)
__global__ void __launch_bounds__(256) attn_kernel(
    const bf16* __restrict__ Qb, const bf16* __restrict__ Kb, const bf16* __restrict__ Vb,
    const bf16* __restrict__ PKg, const bf16* __restrict__ PQg,
    const float* __restrict__ mask, bf16* __restrict__ Ctx, float scale)
{
  const int l0 = blockIdx.x * 16;
  const int bh = blockIdx.y;
  const int bIdx = bh >> 3, h = bh & 7;
  const int wid = threadIdx.x >> 5, lane = threadIdx.x & 31;

  __shared__ bf16 G1[16][436];   // G1[i][t] = q[l0+i] . PKg[l0+t] ; t = (l-m)+399-l0
  __shared__ float S[16][400];   // score rows
  __shared__ bf16 G2s[8][16][36];// per-wave: G2[j][t] = k[m0+j] . PQg[r0+t]
  __shared__ bf16 Pp[16][448];   // probs, zero-padded to 448

  const bf16* Qh  = Qb + (size_t)bIdx * LL * HH + h * HDD;
  const bf16* Kh  = Kb + (size_t)bIdx * LL * HH + h * HDD;
  const bf16* PKh = PKg + (size_t)h * NDELTA * HDD;
  const bf16* PQh = PQg + (size_t)h * NDELTA * HDD;

  v16bf qa0 = frag_load(Qh, HH, l0, 0);
  v16bf qa1 = frag_load(Qh, HH, l0, 32);

  // Stage 1: c2p diagonal table via WMMA (27 column tiles of 16)
  for (int tt = wid; tt < 27; tt += 8) {
    v8f acc = {};
    acc = wmma_bf16(qa0, frag_load_guard(PKh, HDD, l0 + tt * 16, 0,  NDELTA), acc);
    acc = wmma_bf16(qa1, frag_load_guard(PKh, HDD, l0 + tt * 16, 32, NDELTA), acc);
    int tcol = tt * 16 + (lane & 15);
    int ib = (lane >> 4) * 8;
#pragma unroll
    for (int v = 0; v < 8; ++v) G1[ib + v][tcol] = (bf16)acc[v];
  }
  __syncthreads();

  const float* mrow = mask + bIdx * LL;

  // Stage 2: scores = (q.k + c2p + p2c) * scale, masked
  for (int mt = wid; mt < 25; mt += 8) {
    int m0 = mt * 16;
    v16bf ka0 = frag_load(Kh, HH, m0, 0);
    v16bf ka1 = frag_load(Kh, HH, m0, 32);
    v8f accS = {};
    accS = wmma_bf16(qa0, ka0, accS);
    accS = wmma_bf16(qa1, ka1, accS);
    int r0 = 384 + l0 - m0;                    // PQg window start (d = l-m-15 .. l-m+16)
    v8f g2a = {}, g2b = {};
    g2a = wmma_bf16(ka0, frag_load_guard(PQh, HDD, r0,      0,  NDELTA), g2a);
    g2a = wmma_bf16(ka1, frag_load_guard(PQh, HDD, r0,      32, NDELTA), g2a);
    g2b = wmma_bf16(ka0, frag_load_guard(PQh, HDD, r0 + 16, 0,  NDELTA), g2b);
    g2b = wmma_bf16(ka1, frag_load_guard(PQh, HDD, r0 + 16, 32, NDELTA), g2b);
    {
      int t = lane & 15, jb = (lane >> 4) * 8;
#pragma unroll
      for (int v = 0; v < 8; ++v) {
        G2s[wid][jb + v][t]      = (bf16)g2a[v];
        G2s[wid][jb + v][16 + t] = (bf16)g2b[v];
      }
    }
    int j = lane & 15, ib = (lane >> 4) * 8;
#pragma unroll
    for (int v = 0; v < 8; ++v) {
      int i = ib + v, l = l0 + i, m = m0 + j;
      float s = accS[v] + (float)G1[i][i + 399 - m] + (float)G2s[wid][j][i - j + 15];
      s *= scale;
      if (!((mrow[l] != 0.f) && (mrow[m] != 0.f))) s = -3.402823466e38f;
      S[i][m] = s;
    }
  }
  __syncthreads();

  // Stage 3: row softmax, write bf16 probs (zero-padded)
  for (int rr = 0; rr < 2; ++rr) {
    int r = wid + rr * 8;
    float mx = -3.402823466e38f;
    for (int c = lane; c < LL; c += 32) mx = fmaxf(mx, S[r][c]);
#pragma unroll
    for (int o = 16; o; o >>= 1) mx = fmaxf(mx, __shfl_xor(mx, o, 32));
    float sum = 0.f;
    for (int c = lane; c < LL; c += 32) { float e = __expf(S[r][c] - mx); S[r][c] = e; sum += e; }
#pragma unroll
    for (int o = 16; o; o >>= 1) sum += __shfl_xor(sum, o, 32);
    float inv = 1.f / sum;
    for (int c = lane; c < LL; c += 32) Pp[r][c] = (bf16)(S[r][c] * inv);
    for (int c = LL + lane; c < 448; c += 32) Pp[r][c] = (bf16)0.0f;
  }
  __syncthreads();

  // Stage 4: ctx = P @ V (waves 0..3 each own a 16-wide d-tile)
  if (wid < 4) {
    const bf16* Vbase = Vb + (size_t)bIdx * LL * HH;
    int col0 = h * HDD + wid * 16;
    v8f acc = {};
    for (int kt = 0; kt < 14; ++kt)
      acc = wmma_bf16(frag_load(&Pp[0][0], 448, 0, kt * 32),
                      frag_load_t(Vbase, HH, kt * 32, col0, LL), acc);
    int j = lane & 15, ib = (lane >> 4) * 8;
#pragma unroll
    for (int v = 0; v < 8; ++v) {
      int l = l0 + ib + v;
      Ctx[((size_t)(bIdx * LL + l)) * HH + col0 + j] = (bf16)acc[v];
    }
  }
}

// ---------------------------------------------------------------- pooling & heads

__global__ void pool_kernel(const float* __restrict__ outs, const float* __restrict__ masks,
                            float* __restrict__ pooled) {
  int o = blockIdx.x >> 3, b = blockIdx.x & 7;
  int c = threadIdx.x;                         // 256
  const float* Mw = masks + (o * BB + b) * LL;
  const float* X  = outs + ((size_t)o * MM + b * LL) * TT;
  float s = 0.f, wsum = 0.f, mx = -3.402823466e38f;
  for (int l = 0; l < LL; ++l) {
    float wv = Mw[l];
    float xw = X[(size_t)l * TT + c] * wv;
    s += xw; mx = fmaxf(mx, xw); wsum += wv;
  }
  pooled[(size_t)(o * BB + b) * 512 + c]       = s / (wsum + 1e-6f);
  pooled[(size_t)(o * BB + b) * 512 + 256 + c] = mx;
}

__global__ void meanmax_kernel(const float* __restrict__ featsf, float* __restrict__ mf) {
  int b = blockIdx.x, c = threadIdx.x;         // 512
  float s = 0.f, mx = -3.402823466e38f;
  for (int l = 0; l < LL; ++l) {
    float v = featsf[((size_t)(b * LL + l)) * HH + c];
    s += v; mx = fmaxf(mx, v);
  }
  mf[b * 1024 + c]       = s / (float)LL;
  mf[b * 1024 + 512 + c] = mx;
}

__global__ void __launch_bounds__(256) heads_kernel(
    const float* __restrict__ pooled, const float* __restrict__ mf,
    const float* __restrict__ w_bow1, const float* __restrict__ b_bow1,
    const float* __restrict__ w_bow2, const float* __restrict__ b_bow2,
    const float* __restrict__ w_ext1, const float* __restrict__ b_ext1,
    const float* __restrict__ w_ext2, const float* __restrict__ b_ext2,
    const float* __restrict__ w_org1, const float* __restrict__ b_org1,
    const float* __restrict__ w_org2, const float* __restrict__ b_org2,
    float* __restrict__ out)
{
  int b = blockIdx.x, t = threadIdx.x;
  __shared__ float hid[5][256];
  {
    float acc = b_bow1[t];
    const float* p = pooled + (size_t)(0 * BB + b) * 512;
    for (int k = 0; k < 512; ++k) acc += p[k] * w_bow1[k * 256 + t];
    hid[0][t] = mishf(acc);
  }
  {
    float acc = b_ext1[t];
    const float* p = mf + b * 1024;
    for (int k = 0; k < 1024; ++k) acc += p[k] * w_ext1[k * 256 + t];
    hid[1][t] = mishf(acc);
  }
  for (int j = 0; j < 3; ++j) {
    float acc = b_org1[j * 256 + t];
    const float* p = pooled + (size_t)((1 + j) * BB + b) * 512;
    for (int k = 0; k < 512; ++k) acc += p[k] * w_org1[((size_t)j * 512 + k) * 256 + t];
    hid[2 + j][t] = mishf(acc);
  }
  __syncthreads();
  int wid = t >> 5, lane = t & 31;
  for (int oi = wid; oi < 11; oi += 8) {
    const float* hv; const float* wp; int stride; float bb;
    if (oi == 0)      { hv = hid[0]; wp = w_bow2; stride = 1; bb = b_bow2[0]; }
    else if (oi == 1) { hv = hid[1]; wp = w_ext2; stride = 1; bb = b_ext2[0]; }
    else {
      int j = (oi - 2) / 3, c = (oi - 2) % 3;
      hv = hid[2 + j]; wp = w_org2 + j * 768 + c; stride = 3; bb = b_org2[j * 3 + c];
    }
    float s = 0.f;
    for (int k = lane; k < 256; k += 32) s += hv[k] * wp[k * stride];
#pragma unroll
    for (int o = 16; o; o >>= 1) s += __shfl_xor(s, o, 32);
    if (lane == 0) out[b * 11 + oi] = s + bb;
  }
  if (t == 0) out[88 + b] = 0.f;               // second return value: zeros(B)
}

// ---------------------------------------------------------------- launch

extern "C" void kernel_launch(void* const* d_in, const int* in_sizes, int n_in,
                              void* d_out, int out_size, void* d_ws, size_t ws_size,
                              hipStream_t stream) {
  (void)in_sizes; (void)n_in; (void)out_size; (void)ws_size;
  const float* x       = (const float*)d_in[0];
  const float* w_mlp   = (const float*)d_in[1];
  const float* b_mlp   = (const float*)d_in[2];
  const float* w_cnn   = (const float*)d_in[3];
  const float* b_cnn   = (const float*)d_in[4];
  const float* rel_emb = (const float*)d_in[5];
  const float* rel_g   = (const float*)d_in[6];
  const float* rel_b   = (const float*)d_in[7];
  const float* wq      = (const float*)d_in[8];
  const float* wk      = (const float*)d_in[9];
  const float* wv      = (const float*)d_in[10];
  const float* wo      = (const float*)d_in[11];
  const float* wi      = (const float*)d_in[12];
  const float* bq      = (const float*)d_in[13];
  const float* bk      = (const float*)d_in[14];
  const float* bv      = (const float*)d_in[15];
  const float* bo      = (const float*)d_in[16];
  const float* bi      = (const float*)d_in[17];
  const float* wout    = (const float*)d_in[18];
  const float* bout    = (const float*)d_in[19];
  const float* ln1g    = (const float*)d_in[20];
  const float* ln1b    = (const float*)d_in[21];
  const float* ln2g    = (const float*)d_in[22];
  const float* ln2b    = (const float*)d_in[23];
  const float* w_bow1  = (const float*)d_in[24];
  const float* b_bow1  = (const float*)d_in[25];
  const float* w_bow2  = (const float*)d_in[26];
  const float* b_bow2  = (const float*)d_in[27];
  const float* w_ext1  = (const float*)d_in[28];
  const float* b_ext1  = (const float*)d_in[29];
  const float* w_ext2  = (const float*)d_in[30];
  const float* b_ext2  = (const float*)d_in[31];
  const float* w_org1  = (const float*)d_in[32];
  const float* b_org1  = (const float*)d_in[33];
  const float* w_org2  = (const float*)d_in[34];
  const float* b_org2  = (const float*)d_in[35];
  float* out = (float*)d_out;

  char* wsp = (char*)d_ws;
  size_t off = 0;
  auto alloc = [&](size_t bytes) -> void* {
    void* p = wsp + off;
    off = (off + bytes + 255) & ~(size_t)255;
    return p;
  };

  bf16* xb      = (bf16*)alloc((size_t)MM * FF * 2);
  bf16* featsb  = (bf16*)alloc((size_t)MM * HH * 2);
  float* featsf = (float*)alloc((size_t)MM * HH * 4);
  float* masks  = (float*)alloc((size_t)4 * BB * LL * 4);
  int*  c2pT    = (int*)alloc(NDELTA * 4);
  int*  p2cT    = (int*)alloc(NDELTA * 4);
  bf16* wt_mlp  = (bf16*)alloc((size_t)256 * 128 * 2);
  bf16* wt_cnn  = (bf16*)alloc((size_t)256 * 768 * 2);
  bf16* wtq     = (bf16*)alloc((size_t)4 * 512 * 512 * 2);
  bf16* wtk     = (bf16*)alloc((size_t)4 * 512 * 512 * 2);
  bf16* wtv     = (bf16*)alloc((size_t)4 * 512 * 512 * 2);
  bf16* wto_    = (bf16*)alloc((size_t)4 * 512 * 512 * 2);
  bf16* wti     = (bf16*)alloc((size_t)4 * 512 * 512 * 2);
  bf16* wtout   = (bf16*)alloc((size_t)4 * 256 * 512 * 2);
  bf16* relb    = (bf16*)alloc((size_t)PPOS * HH * 2);
  bf16* qbuf    = (bf16*)alloc((size_t)MM * HH * 2);
  bf16* kbuf    = (bf16*)alloc((size_t)MM * HH * 2);
  bf16* vbuf    = (bf16*)alloc((size_t)MM * HH * 2);
  bf16* pkbuf   = (bf16*)alloc((size_t)PPOS * HH * 2);
  bf16* pqbuf   = (bf16*)alloc((size_t)PPOS * HH * 2);
  bf16* PKg     = (bf16*)alloc((size_t)NHH * NDELTA * HDD * 2);
  bf16* PQg     = (bf16*)alloc((size_t)NHH * NDELTA * HDD * 2);
  bf16* ctxb    = (bf16*)alloc((size_t)MM * HH * 2);
  float* ubuf   = (float*)alloc((size_t)MM * HH * 4);
  bf16* h1b     = (bf16*)alloc((size_t)MM * HH * 2);
  bf16* interb  = (bf16*)alloc((size_t)MM * HH * 2);
  float* v2     = (float*)alloc((size_t)MM * TT * 4);
  float* outs   = (float*)alloc((size_t)4 * MM * TT * 4);
  float* pooled = (float*)alloc((size_t)4 * BB * 512 * 4);
  float* mf     = (float*)alloc((size_t)BB * 1024 * 4);

  const float scale = 0.07216878364870323f;    // 1/sqrt(64*3)

  prep_kernel<<<(MM * FF + 255) / 256, 256, 0, stream>>>(x, xb, masks);
  idx_kernel<<<4, 256, 0, stream>>>(c2pT, p2cT);
  pack_wt<<<(128 * 256 + 255) / 256, 256, 0, stream>>>(w_mlp, wt_mlp, 128, 256);
  pack_wt<<<(768 * 256 + 255) / 256, 256, 0, stream>>>(w_cnn, wt_cnn, 768, 256);
  for (int o = 0; o < 4; ++o) {
    size_t wOff = (size_t)o * 512 * 512;
    pack_wt<<<1024, 256, 0, stream>>>(wq + wOff, wtq  + wOff, 512, 512);
    pack_wt<<<1024, 256, 0, stream>>>(wk + wOff, wtk  + wOff, 512, 512);
    pack_wt<<<1024, 256, 0, stream>>>(wv + wOff, wtv  + wOff, 512, 512);
    pack_wt<<<1024, 256, 0, stream>>>(wo + wOff, wto_ + wOff, 512, 512);
    pack_wt<<<1024, 256, 0, stream>>>(wi + wOff, wti  + wOff, 512, 512);
    pack_wt<<<512, 256, 0, stream>>>(wout + (size_t)o * 512 * 256, wtout + (size_t)o * 256 * 512, 512, 256);
  }

  // front end: f1 = mish(x@Wmlp+b), f2 = relu(conv3(f1)) -> feats = [f1 | f2]
  gemm_wmma<0, 1><<<dim3(50, 2), 256, 0, stream>>>(xb, FF, wt_mlp, b_mlp, nullptr, 0,
      featsf, HH, featsb, HH, MM, 256, 128);
  gemm_wmma<1, 2><<<dim3(50, 2), 256, 0, stream>>>(featsb, HH, wt_cnn, b_cnn, nullptr, 0,
      featsf + 256, HH, featsb + 256, HH, MM, 256, 768);

  for (int o = 0; o < 4; ++o) {
    size_t wOff = (size_t)o * 512 * 512;
    ln_kernel<<<PPOS, 256, 0, stream>>>(rel_emb + (size_t)o * PPOS * HH, rel_g + o * HH, rel_b + o * HH,
                                        nullptr, relb, HH);
    gemm_wmma<0, 0><<<dim3(50, 4), 256, 0, stream>>>(featsb, HH, wtq + wOff, bq + o * HH, nullptr, 0,
        nullptr, 0, qbuf, HH, MM, HH, HH);
    gemm_wmma<0, 0><<<dim3(50, 4), 256, 0, stream>>>(featsb, HH, wtk + wOff, bk + o * HH, nullptr, 0,
        nullptr, 0, kbuf, HH, MM, HH, HH);
    gemm_wmma<0, 0><<<dim3(50, 4), 256, 0, stream>>>(featsb, HH, wtv + wOff, bv + o * HH, nullptr, 0,
        nullptr, 0, vbuf, HH, MM, HH, HH);
    gemm_wmma<0, 0><<<dim3(19, 4), 256, 0, stream>>>(relb, HH, wtq + wOff, bq + o * HH, nullptr, 0,
        nullptr, 0, pqbuf, HH, PPOS, HH, HH);
    gemm_wmma<0, 0><<<dim3(19, 4), 256, 0, stream>>>(relb, HH, wtk + wOff, bk + o * HH, nullptr, 0,
        nullptr, 0, pkbuf, HH, PPOS, HH, HH);
    gather_pos<<<(NHH * NDELTA * HDD + 255) / 256, 256, 0, stream>>>(pkbuf, pqbuf, c2pT, p2cT, PKg, PQg);
    attn_kernel<<<dim3(25, BB * NHH), 256, 0, stream>>>(qbuf, kbuf, vbuf, PKg, PQg,
        masks + (size_t)o * BB * LL, ctxb, scale);
    gemm_wmma<0, 0><<<dim3(50, 4), 256, 0, stream>>>(ctxb, HH, wto_ + wOff, bo + o * HH, featsf, HH,
        ubuf, HH, nullptr, 0, MM, HH, HH);
    ln_kernel<<<MM, 256, 0, stream>>>(ubuf, ln1g + o * HH, ln1b + o * HH, nullptr, h1b, HH);
    gemm_wmma<0, 1><<<dim3(50, 4), 256, 0, stream>>>(h1b, HH, wti + wOff, bi + o * HH, nullptr, 0,
        nullptr, 0, interb, HH, MM, HH, HH);
    gemm_wmma<0, 0><<<dim3(50, 2), 256, 0, stream>>>(interb, HH, wtout + (size_t)o * 256 * 512, bout + o * TT,
        nullptr, 0, v2, TT, nullptr, 0, MM, TT, HH);
    ln_kernel<<<MM, 256, 0, stream>>>(v2, ln2g + o * TT, ln2b + o * TT,
        outs + (size_t)o * MM * TT, nullptr, TT);
  }

  pool_kernel<<<32, 256, 0, stream>>>(outs, masks, pooled);
  meanmax_kernel<<<BB, 512, 0, stream>>>(featsf, mf);
  heads_kernel<<<BB, 256, 0, stream>>>(pooled, mf, w_bow1, b_bow1, w_bow2, b_bow2,
      w_ext1, b_ext1, w_ext2, b_ext2, w_org1, b_org1, w_org2, b_org2, out);
}
